// VoxelProposal_51161650430130
// MI455X (gfx1250) — matrix operand order repaired and checked by reading, compiled, not run on hardware
//
#include <hip/hip_runtime.h>

typedef __attribute__((ext_vector_type(16))) _Float16 v16h;
typedef __attribute__((ext_vector_type(8)))  _Float16 v8h;
typedef __attribute__((ext_vector_type(8)))  float    v8f;

#define N_VOX   131072
#define C_DIM   128
#define TILE_V  32

// workspace layout (bytes)
#define OFF_INV     0           // 25 floats (invK 9 + invE 16)
#define OFF_MASK    256         // 131072 bytes
#define OFF_WPACK   131328      // 26 ntiles * 4 kb * 32 lanes * 16 halves = 106496 B
#define OFF_WVPACK  237824      // 8 * 4 * 32 * 16 halves = 32768 B
#define OFF_VALUE   270592      // 10080 * 128 f32 = 5160960 B

// ---------------------------------------------------------------- helpers
__device__ __forceinline__ v16h make_a_frag(const _Float16* p) {
  // A 16x32 f16 layout: this lane needs K = {8*kh+0..7} and {16+8*kh+0..7};
  // caller passes p = row_base + kb*32 + 8*kh  (16B aligned)
  v8h lo = *(const v8h*)(p);
  v8h hi = *(const v8h*)(p + 16);
  v16h a;
#pragma unroll
  for (int j = 0; j < 8; ++j) { a[j] = lo[j]; a[j + 8] = hi[j]; }
  return a;
}

__device__ void inv3(const float* m, float* o) {
  float a=m[0],b=m[1],c=m[2],d=m[3],e=m[4],f=m[5],g=m[6],h=m[7],i=m[8];
  float A =  (e*i - f*h), B = -(d*i - f*g), C =  (d*h - e*g);
  float det = a*A + b*B + c*C;
  float id = 1.0f/det;
  o[0]=A*id;            o[1]=-(b*i-c*h)*id;  o[2]=(b*f-c*e)*id;
  o[3]=B*id;            o[4]=(a*i-c*g)*id;   o[5]=-(a*f-c*d)*id;
  o[6]=C*id;            o[7]=-(a*h-b*g)*id;  o[8]=(a*e-b*d)*id;
}

__device__ void inv4(const float* m, float* o) {
  float inv[16];
  inv[0]= m[5]*m[10]*m[15]-m[5]*m[11]*m[14]-m[9]*m[6]*m[15]+m[9]*m[7]*m[14]+m[13]*m[6]*m[11]-m[13]*m[7]*m[10];
  inv[4]=-m[4]*m[10]*m[15]+m[4]*m[11]*m[14]+m[8]*m[6]*m[15]-m[8]*m[7]*m[14]-m[12]*m[6]*m[11]+m[12]*m[7]*m[10];
  inv[8]= m[4]*m[9]*m[15]-m[4]*m[11]*m[13]-m[8]*m[5]*m[15]+m[8]*m[7]*m[13]+m[12]*m[5]*m[11]-m[12]*m[7]*m[9];
  inv[12]=-m[4]*m[9]*m[14]+m[4]*m[10]*m[13]+m[8]*m[5]*m[14]-m[8]*m[6]*m[13]-m[12]*m[5]*m[10]+m[12]*m[6]*m[9];
  inv[1]=-m[1]*m[10]*m[15]+m[1]*m[11]*m[14]+m[9]*m[2]*m[15]-m[9]*m[3]*m[14]-m[13]*m[2]*m[11]+m[13]*m[3]*m[10];
  inv[5]= m[0]*m[10]*m[15]-m[0]*m[11]*m[14]-m[8]*m[2]*m[15]+m[8]*m[3]*m[14]+m[12]*m[2]*m[11]-m[12]*m[3]*m[10];
  inv[9]=-m[0]*m[9]*m[15]+m[0]*m[11]*m[13]+m[8]*m[1]*m[15]-m[8]*m[3]*m[13]-m[12]*m[1]*m[11]+m[12]*m[3]*m[9];
  inv[13]= m[0]*m[9]*m[14]-m[0]*m[10]*m[13]-m[8]*m[1]*m[14]+m[8]*m[2]*m[13]+m[12]*m[1]*m[10]-m[12]*m[2]*m[9];
  inv[2]= m[1]*m[6]*m[15]-m[1]*m[7]*m[14]-m[5]*m[2]*m[15]+m[5]*m[3]*m[14]+m[13]*m[2]*m[7]-m[13]*m[3]*m[6];
  inv[6]=-m[0]*m[6]*m[15]+m[0]*m[7]*m[14]+m[4]*m[2]*m[15]-m[4]*m[3]*m[14]-m[12]*m[2]*m[7]+m[12]*m[3]*m[6];
  inv[10]= m[0]*m[5]*m[15]-m[0]*m[7]*m[13]-m[4]*m[1]*m[15]+m[4]*m[3]*m[13]+m[12]*m[1]*m[7]-m[12]*m[3]*m[5];
  inv[14]=-m[0]*m[5]*m[14]+m[0]*m[6]*m[13]+m[4]*m[1]*m[14]-m[4]*m[2]*m[13]-m[12]*m[1]*m[6]+m[12]*m[2]*m[5];
  inv[3]=-m[1]*m[6]*m[11]+m[1]*m[7]*m[10]+m[5]*m[2]*m[11]-m[5]*m[3]*m[10]-m[9]*m[2]*m[7]+m[9]*m[3]*m[6];
  inv[7]= m[0]*m[6]*m[11]-m[0]*m[7]*m[10]-m[4]*m[2]*m[11]+m[4]*m[3]*m[10]+m[8]*m[2]*m[7]-m[8]*m[3]*m[6];
  inv[11]=-m[0]*m[5]*m[11]+m[0]*m[7]*m[9]+m[4]*m[1]*m[11]-m[4]*m[3]*m[9]-m[8]*m[1]*m[7]+m[8]*m[3]*m[5];
  inv[15]= m[0]*m[5]*m[10]-m[0]*m[6]*m[9]-m[4]*m[1]*m[10]+m[4]*m[2]*m[9]+m[8]*m[1]*m[6]-m[8]*m[2]*m[5];
  float det = m[0]*inv[0]+m[1]*inv[4]+m[2]*inv[8]+m[3]*inv[12];
  det = 1.0f/det;
  for (int i=0;i<16;++i) o[i]=inv[i]*det;
}

// ---------------------------------------------------------------- kernels
__global__ void k_invert(const float* K, const float* E, float* ws) {
  if (threadIdx.x == 0) { inv3(K, ws); inv4(E, ws + 9); }
}

// Pack weights into f16 WMMA B-fragment order:
// B 32x16 f16 layout: lane n = l%16, kh = l/16; half j -> K = kb*32 + 16*(j/8) + 8*kh + j%8
__global__ __launch_bounds__(256) void k_pack(
    const float* Woff, const float* Wattw, const float* Wout, const float* Wv,
    _Float16* wpack, _Float16* wvpack) {
  int id = blockIdx.x * 256 + threadIdx.x;
  if (id < 53248) {
    int j  = id & 15;
    int l  = (id >> 4) & 31;
    int kb = (id >> 9) & 3;
    int nt = id >> 11;               // 0..25
    int n  = nt * 16 + (l & 15);
    int K  = kb * 32 + (j >> 3) * 16 + (l >> 4) * 8 + (j & 7);
    float s;
    if (n < 192)      s = Woff[n * 128 + K];
    else if (n < 288) s = Wattw[(n - 192) * 128 + K];
    else              s = Wout[(n - 288) * 128 + K];
    wpack[id] = (_Float16)s;
  } else if (id < 69632) {
    int id2 = id - 53248;
    int j  = id2 & 15;
    int l  = (id2 >> 4) & 31;
    int kb = (id2 >> 9) & 3;
    int nt = id2 >> 11;              // 0..7
    int n  = nt * 16 + (l & 15);
    int K  = kb * 32 + (j >> 3) * 16 + (l >> 4) * 8 + (j & 7);
    wvpack[id2] = (_Float16)Wv[n * 128 + K];
  }
}

__global__ __launch_bounds__(256) void k_zero(unsigned int* mask32) {
  int i = blockIdx.x * 256 + threadIdx.x;
  if (i < N_VOX / 4) mask32[i] = 0u;
}

__global__ __launch_bounds__(256) void k_mask(
    const float* depth, const float* ws, const float* vorg, unsigned char* mask) {
  int idx = blockIdx.x * 256 + threadIdx.x;
  if (idx >= 192 * 640) return;
  int px = idx % 640, py = idx / 640;
  float d = depth[idx];
  float x0 = (float)px * d, x1 = (float)py * d, x2 = d;
  const float* iK = ws; const float* iE = ws + 9;
  float c0 = iK[0]*x0 + iK[1]*x1 + iK[2]*x2;
  float c1 = iK[3]*x0 + iK[4]*x1 + iK[5]*x2;
  float c2 = iK[6]*x0 + iK[7]*x1 + iK[8]*x2;
  float w0 = iE[0]*c0 + iE[1]*c1 + iE[2]*c2  + iE[3];
  float w1 = iE[4]*c0 + iE[5]*c1 + iE[6]*c2  + iE[7];
  float w2 = iE[8]*c0 + iE[9]*c1 + iE[10]*c2 + iE[11];
  float vx = (w0 - vorg[0]) / 0.4f - 0.5f;
  float vy = (w1 - vorg[1]) / 0.4f - 0.5f;
  float vz = ((w2 - vorg[2]) / 0.4f - 0.5f) / 2.0f;
  vx = truncf(vx); vy = truncf(vy); vz = truncf(vz);
  if (vx >= 0.f && vx < 128.f && vy >= 0.f && vy < 128.f && vz >= 0.f && vz < 8.f) {
    int ix = (int)vx, iy = (int)vy, iz = (int)vz;
    mask[(ix * 128 + iy) * 8 + iz] = (unsigned char)1;
  }
}

// value[p][c'] = sum_c feat[c][p] * Wv[c'][c] + bv[c']   (f16 WMMA, f32 accum)
__global__ __launch_bounds__(256) void k_value(
    const float* feat0, const float* feat1, const float* feat2,
    const _Float16* wvpack, const float* bv, float* value) {
  int mt   = blockIdx.x;           // 0..629
  int tid  = threadIdx.x;
  int lane = tid & 31;
  int nt   = tid >> 5;             // 0..7
  int m16  = lane & 15;
  int kh   = lane >> 4;

  const float* f; int HW, pl, pg;
  if (mt < 480)      { f = feat0; HW = 7680; pl = mt * 16;         pg = pl; }
  else if (mt < 600) { f = feat1; HW = 1920; pl = (mt - 480) * 16; pg = 7680 + pl; }
  else               { f = feat2; HW = 480;  pl = (mt - 600) * 16; pg = 9600 + pl; }

  v8f acc = {};
#pragma unroll
  for (int kb = 0; kb < 4; ++kb) {
    v16h A;
#pragma unroll
    for (int j = 0; j < 8; ++j) {
      int c  = kb * 32 + 8 * kh + j;
      A[j]     = (_Float16)f[c * HW + pl + m16];
      A[j + 8] = (_Float16)f[(c + 16) * HW + pl + m16];
    }
    v16h B = *(const v16h*)(wvpack + ((size_t)(nt * 4 + kb) * 32 + lane) * 16);
    acc = __builtin_amdgcn_wmma_f32_16x16x32_f16(false, A, false, B, (short)0, acc,
                                                 false, false);
  }
  int col = nt * 16 + m16;
  float b = bv[col];
#pragma unroll
  for (int v = 0; v < 8; ++v) {
    int row = pg + v + 8 * kh;
    value[(size_t)row * 128 + col] = acc[v] + b;
  }
}

// fused: off/attw proj -> softmax -> bilinear gather -> out proj -> LN -> masked write
__global__ __launch_bounds__(256) void k_main(
    const float* __restrict__ q, const float* __restrict__ refpix,
    const float* __restrict__ value, const _Float16* __restrict__ wpack,
    const float* __restrict__ boff, const float* __restrict__ battw,
    const float* __restrict__ bout, const float* __restrict__ lng,
    const float* __restrict__ lnb, const unsigned char* __restrict__ mask,
    float* __restrict__ out) {
  __shared__ __align__(16) _Float16 s_qh[TILE_V * C_DIM];      // 8 KB
  __shared__ __align__(16) float    s_proj[TILE_V * 288];      // 36 KB (reused as out2)
  __shared__ __align__(16) _Float16 s_samph[TILE_V * C_DIM];   // 8 KB
  __shared__ float s_red[TILE_V * 8 * 2];                      // 2 KB

  const int tid  = threadIdx.x;
  const int lane = tid & 31;
  const int wid  = tid >> 5;
  const int m16  = lane & 15;
  const int kh   = lane >> 4;
  const int n0   = blockIdx.x * TILE_V;

  // ---- stage 1: q tile -> f16 LDS
  for (int i = tid; i < TILE_V * C_DIM; i += 256)
    s_qh[i] = (_Float16)q[(size_t)n0 * C_DIM + i];
  __syncthreads();

  // ---- stage 2: proj = q @ [Woff;Wattw]^T + bias   (32x288, f16 WMMA)
  for (int job = wid; job < 36; job += 8) {
    int mt = job & 1, nt = job >> 1;          // nt 0..17
    v8f acc = {};
#pragma unroll
    for (int kb = 0; kb < 4; ++kb) {
      v16h A = make_a_frag(&s_qh[(mt * 16 + m16) * C_DIM + kb * 32 + 8 * kh]);
      v16h B = *(const v16h*)(wpack + ((size_t)(nt * 4 + kb) * 32 + lane) * 16);
      acc = __builtin_amdgcn_wmma_f32_16x16x32_f16(false, A, false, B, (short)0, acc,
                                                   false, false);
    }
    int col = nt * 16 + m16;
    float b = (col < 192) ? boff[col] : battw[col - 192];
#pragma unroll
    for (int v = 0; v < 8; ++v)
      s_proj[(mt * 16 + v + 8 * kh) * 288 + col] = acc[v] + b;
  }
  __syncthreads();

  // ---- stage 3: softmax + bilinear sampling (one (voxel,head) pair per thread)
  {
    int v = tid >> 3, h = tid & 7;
    const float* prow = &s_proj[v * 288];
    float w[12];
    float mx = -1e30f;
#pragma unroll
    for (int t = 0; t < 12; ++t) { w[t] = prow[192 + h * 12 + t]; mx = fmaxf(mx, w[t]); }
    float s = 0.f;
#pragma unroll
    for (int t = 0; t < 12; ++t) { w[t] = expf(w[t] - mx); s += w[t]; }
#pragma unroll
    for (int t = 0; t < 12; ++t) w[t] /= s;

    float rx = refpix[(size_t)(n0 + v) * 2 + 0];
    float ry = refpix[(size_t)(n0 + v) * 2 + 1];
    float acc[16];
#pragma unroll
    for (int i = 0; i < 16; ++i) acc[i] = 0.f;

    const int Hs[3] = {48, 24, 12}, Wls[3] = {160, 80, 40}, Ss[3] = {0, 7680, 9600};
#pragma unroll
    for (int lvl = 0; lvl < 3; ++lvl) {
      const int Hl = Hs[lvl], Wl = Wls[lvl], st = Ss[lvl];
#pragma unroll
      for (int pt = 0; pt < 4; ++pt) {
        float ox = prow[h * 24 + lvl * 8 + pt * 2 + 0];
        float oy = prow[h * 24 + lvl * 8 + pt * 2 + 1];
        float x  = (rx + ox / (float)Wl) * (float)Wl - 0.5f;
        float y  = (ry + oy / (float)Hl) * (float)Hl - 0.5f;
        float x0f = floorf(x), y0f = floorf(y);
        float lx = x - x0f, ly = y - y0f;
        int x0 = (int)x0f, y0 = (int)y0f;
        float ww = w[lvl * 4 + pt];
        float tw[4] = {(1.f - lx) * (1.f - ly), lx * (1.f - ly),
                       (1.f - lx) * ly,         lx * ly};
        int xs[4] = {x0, x0 + 1, x0, x0 + 1};
        int ys[4] = {y0, y0, y0 + 1, y0 + 1};
#pragma unroll
        for (int tt = 0; tt < 4; ++tt) {
          int xi = xs[tt], yi = ys[tt];
          if (xi >= 0 && xi < Wl && yi >= 0 && yi < Hl) {
            const float4* g = reinterpret_cast<const float4*>(
                value + (size_t)(st + yi * Wl + xi) * 128 + h * 16);
            float wt = tw[tt] * ww;
#pragma unroll
            for (int kq = 0; kq < 4; ++kq) {
              float4 gv = g[kq];
              acc[kq * 4 + 0] += gv.x * wt;
              acc[kq * 4 + 1] += gv.y * wt;
              acc[kq * 4 + 2] += gv.z * wt;
              acc[kq * 4 + 3] += gv.w * wt;
            }
          }
        }
      }
    }
    _Float16* srow = &s_samph[v * 128 + h * 16];
#pragma unroll
    for (int i = 0; i < 16; ++i) srow[i] = (_Float16)acc[i];
  }
  __syncthreads();

  // ---- stage 4: out2 = samp @ Wout^T + bout  (reuse s_proj as [32][128] f32)
  for (int job = wid; job < 16; job += 8) {
    int mt = job & 1, nt = job >> 1;          // nt 0..7 -> wpack tiles 18..25
    v8f acc = {};
#pragma unroll
    for (int kb = 0; kb < 4; ++kb) {
      v16h A = make_a_frag(&s_samph[(mt * 16 + m16) * C_DIM + kb * 32 + 8 * kh]);
      v16h B = *(const v16h*)(wpack + ((size_t)((18 + nt) * 4 + kb) * 32 + lane) * 16);
      acc = __builtin_amdgcn_wmma_f32_16x16x32_f16(false, A, false, B, (short)0, acc,
                                                   false, false);
    }
    int col = nt * 16 + m16;
    float b = bout[col];
#pragma unroll
    for (int v = 0; v < 8; ++v)
      s_proj[(mt * 16 + v + 8 * kh) * 128 + col] = acc[v] + b;
  }
  __syncthreads();

  // ---- stage 5: residual + LayerNorm + mask + transposed write
  {
    int v = tid >> 3, part = tid & 7;
    int c0 = part * 16;
    float hv[16];
    float s1 = 0.f, s2 = 0.f;
#pragma unroll
    for (int ci = 0; ci < 16; ++ci) {
      int c = c0 + ci;
      float hq = q[(size_t)(n0 + v) * C_DIM + c] + s_proj[v * 128 + c];
      hv[ci] = hq;
      s1 += hq; s2 += hq * hq;
    }
    s_red[(v * 8 + part) * 2 + 0] = s1;
    s_red[(v * 8 + part) * 2 + 1] = s2;
    __syncthreads();
    float S1 = 0.f, S2 = 0.f;
#pragma unroll
    for (int p = 0; p < 8; ++p) {
      S1 += s_red[(v * 8 + p) * 2 + 0];
      S2 += s_red[(v * 8 + p) * 2 + 1];
    }
    float mu  = S1 * (1.f / 128.f);
    float var = S2 * (1.f / 128.f) - mu * mu;
    float rstd = rsqrtf(var + 1e-5f);
    bool mk = mask[n0 + v] != 0;
#pragma unroll
    for (int ci = 0; ci < 16; ++ci) {
      int c = c0 + ci;
      float hq = hv[ci];
      float y  = (hq - mu) * rstd * lng[c] + lnb[c];
      float qv = hq - s_proj[v * 128 + c];
      out[(size_t)c * N_VOX + (n0 + v)] = mk ? y : qv;
    }
  }
}

// ---------------------------------------------------------------- launcher
extern "C" void kernel_launch(void* const* d_in, const int* in_sizes, int n_in,
                              void* d_out, int out_size, void* d_ws, size_t ws_size,
                              hipStream_t stream) {
  const float* q      = (const float*)d_in[0];
  const float* feat0  = (const float*)d_in[1];
  const float* feat1  = (const float*)d_in[2];
  const float* feat2  = (const float*)d_in[3];
  const float* depth  = (const float*)d_in[4];
  const float* Kmat   = (const float*)d_in[5];
  const float* Emat   = (const float*)d_in[6];
  const float* vorg   = (const float*)d_in[7];
  const float* refpix = (const float*)d_in[8];
  const float* Wv     = (const float*)d_in[9];
  const float* bv     = (const float*)d_in[10];
  const float* Woff   = (const float*)d_in[11];
  const float* boff   = (const float*)d_in[12];
  const float* Wattw  = (const float*)d_in[13];
  const float* battw  = (const float*)d_in[14];
  const float* Wout   = (const float*)d_in[15];
  const float* bout   = (const float*)d_in[16];
  const float* lng    = (const float*)d_in[17];
  const float* lnb    = (const float*)d_in[18];

  char* ws = (char*)d_ws;
  float*          inv    = (float*)(ws + OFF_INV);
  unsigned char*  mask   = (unsigned char*)(ws + OFF_MASK);
  _Float16*       wpack  = (_Float16*)(ws + OFF_WPACK);
  _Float16*       wvpack = (_Float16*)(ws + OFF_WVPACK);
  float*          value  = (float*)(ws + OFF_VALUE);

  k_invert<<<1, 32, 0, stream>>>(Kmat, Emat, inv);
  k_pack<<<272, 256, 0, stream>>>(Woff, Wattw, Wout, Wv, wpack, wvpack);
  k_zero<<<128, 256, 0, stream>>>((unsigned int*)mask);
  k_mask<<<480, 256, 0, stream>>>(depth, inv, vorg, mask);
  k_value<<<630, 256, 0, stream>>>(feat0, feat1, feat2, wvpack, bv, value);
  k_main<<<N_VOX / TILE_V, 256, 0, stream>>>(q, refpix, value, wpack, boff, battw,
                                             bout, lng, lnb, mask, (float*)d_out);
}
